// _GroundingDinoEncoder_72808285602309
// MI455X (gfx1250) — compile-verified, matrix operand order checked by reading, not run on hardware
//
#include <hip/hip_runtime.h>
#include <math.h>

// ---------------------------------------------------------------------------
// GroundingDino fusion-encoder layer for MI455X (gfx1250, wave32, WMMA).
// All GEMMs go through a bf16 WMMA kernel (v_wmma_f32_16x16x32_bf16) with
// fused bias / ReLU / alpha-scale epilogues and grid.z head-batching.
// Round 2: 64x128 block tile (4 wmma/wave/k-step), vectorized float4->bf16
// LDS staging fast path (no per-element exec masking on interior tiles).
// ---------------------------------------------------------------------------

#define D_MODEL   256
#define SV_CONST  9382          // 84*84 + 42*42 + 21*21 + 11*11
#define ST_CONST  256
#define FP16_MIN  (-65504.0f)

typedef __attribute__((ext_vector_type(8)))  float        v8f;
typedef __attribute__((ext_vector_type(16))) __bf16       v16bf;
typedef __attribute__((ext_vector_type(8)))  unsigned int v8u;

union FragAB { v8u u; v16bf b; };

__device__ __forceinline__ unsigned short f2bf(float f) {
  unsigned int u = __float_as_uint(f);
  unsigned int r = u + 0x7FFFu + ((u >> 16) & 1u);   // round-to-nearest-even
  return (unsigned short)(r >> 16);
}
__device__ __forceinline__ unsigned int pack2(unsigned short lo, unsigned short hi) {
  return (unsigned int)lo | ((unsigned int)hi << 16);
}
__device__ __forceinline__ unsigned int pack2f(float a, float b) {
  return pack2(f2bf(a), f2bf(b));
}

// ------------------------------ block reductions (wave32) -------------------
__device__ __forceinline__ float blk_sum(float v) {
  __shared__ float r[8];
  #pragma unroll
  for (int o = 16; o > 0; o >>= 1) v += __shfl_xor(v, o, 32);
  if ((threadIdx.x & 31) == 0) r[threadIdx.x >> 5] = v;
  __syncthreads();
  float t = r[0];
  #pragma unroll
  for (int i = 1; i < 8; ++i) t += r[i];
  __syncthreads();
  return t;
}
__device__ __forceinline__ float blk_max(float v) {
  __shared__ float r[8];
  #pragma unroll
  for (int o = 16; o > 0; o >>= 1) v = fmaxf(v, __shfl_xor(v, o, 32));
  if ((threadIdx.x & 31) == 0) r[threadIdx.x >> 5] = v;
  __syncthreads();
  float t = r[0];
  #pragma unroll
  for (int i = 1; i < 8; ++i) t = fmaxf(t, r[i]);
  __syncthreads();
  return t;
}

// ------------------------------ WMMA GEMM -----------------------------------
// C[M,N] = epilogue( alpha * (A[M,K] * B + bias) ), B row-major KxN or
// (TRANSB) NxK.  Block tile 64x128, 8 waves (4 row strips x 2 col groups),
// each wave computes 16x64 via 4 f32 accumulators; K step 32.
#define GEMM_BIAS   1
#define GEMM_RELU   2
#define GEMM_TRANSB 4

__global__ __launch_bounds__(256) void gemm_bf16_wmma(
    const float* __restrict__ A, const float* __restrict__ B,
    const float* __restrict__ bias, float* __restrict__ C,
    int M, int N, int K, int lda, int ldb, int ldc,
    long strideA, long strideB, long strideC,
    float alpha, int flags)
{
  __shared__ unsigned short sA[64][40];    // [m][k]   (row stride 80B, 8B-aligned)
  __shared__ unsigned short sB[128][40];   // [n][k]   (B staged n-major)

  A += (long)blockIdx.z * strideA;
  B += (long)blockIdx.z * strideB;
  C += (long)blockIdx.z * strideC;

  const int m0   = blockIdx.y * 64;
  const int n0   = blockIdx.x * 128;
  const int tid  = threadIdx.x;
  const int wave = tid >> 5;
  const int lane = tid & 31;
  const int wr   = (wave & 3) * 16;   // wave row offset inside tile
  const int wc   = (wave >> 2) * 64;  // wave col offset inside tile
  const int half = lane >> 4;         // 0: K base 0/16, 1: K base 8/24
  const int l16  = lane & 15;
  const int kb   = half * 8;

  const bool trB    = (flags & GEMM_TRANSB) != 0;
  const bool aAlign = ((lda & 3) == 0);
  const bool bAlign = ((ldb & 3) == 0);
  const bool fullM  = (m0 + 64 <= M);
  const bool fullN  = (n0 + 128 <= N);

  v8f acc[4] = {v8f{}, v8f{}, v8f{}, v8f{}};

  for (int k0 = 0; k0 < K; k0 += 32) {
    const bool fullK = (k0 + 32 <= K);

    // ---- stage A tile (64x32) as bf16 ----
    if (fullM && fullK && aAlign) {
      #pragma unroll
      for (int t = 0; t < 2; ++t) {
        int i  = tid + t * 256;           // 0..511 float4s
        int r  = i >> 3;
        int kq = (i & 7) * 4;
        float4 f = *(const float4*)(A + (long)(m0 + r) * lda + (k0 + kq));
        uint2 p; p.x = pack2f(f.x, f.y); p.y = pack2f(f.z, f.w);
        *(uint2*)&sA[r][kq] = p;
      }
    } else {
      for (int i = tid; i < 64 * 32; i += 256) {
        int r = i >> 5, kk = i & 31;
        int gm = m0 + r, gk = k0 + kk;
        float f = (gm < M && gk < K) ? A[(long)gm * lda + gk] : 0.0f;
        sA[r][kk] = f2bf(f);
      }
    }

    // ---- stage B tile (32k x 128n) into sB[n][k] ----
    if (trB) {
      if (fullN && fullK && bAlign) {
        #pragma unroll
        for (int t = 0; t < 4; ++t) {
          int i  = tid + t * 256;         // 0..1023 float4s along k
          int n  = i >> 3;
          int kq = (i & 7) * 4;
          float4 f = *(const float4*)(B + (long)(n0 + n) * ldb + (k0 + kq));
          uint2 p; p.x = pack2f(f.x, f.y); p.y = pack2f(f.z, f.w);
          *(uint2*)&sB[n][kq] = p;
        }
      } else {
        for (int i = tid; i < 128 * 32; i += 256) {
          int n = i >> 5, kk = i & 31;
          int gn = n0 + n, gk = k0 + kk;
          float f = (gn < N && gk < K) ? B[(long)gn * ldb + gk] : 0.0f;
          sB[n][kk] = f2bf(f);
        }
      }
    } else {
      if (fullN && fullK && bAlign) {
        #pragma unroll
        for (int t = 0; t < 4; ++t) {
          int i  = tid + t * 256;         // 0..1023 float4s along n
          int kk = i >> 5;
          int nq = (i & 31) * 4;
          float4 f = *(const float4*)(B + (long)(k0 + kk) * ldb + (n0 + nq));
          sB[nq + 0][kk] = f2bf(f.x);
          sB[nq + 1][kk] = f2bf(f.y);
          sB[nq + 2][kk] = f2bf(f.z);
          sB[nq + 3][kk] = f2bf(f.w);
        }
      } else {
        for (int i = tid; i < 128 * 32; i += 256) {
          int n = i & 127, kk = i >> 7;
          int gn = n0 + n, gk = k0 + kk;
          float f = (gn < N && gk < K) ? B[(long)gk * ldb + gn] : 0.0f;
          sB[n][kk] = f2bf(f);
        }
      }
    }
    __syncthreads();

    // ---- fragments per CDNA5 16-bit A/B layouts; A reused across 4 wmma ----
    FragAB fa;
    const int ar = wr + l16;
    #pragma unroll
    for (int j = 0; j < 4; ++j) {
      fa.u[j]     = pack2(sA[ar][kb + 2*j],      sA[ar][kb + 2*j + 1]);
      fa.u[4 + j] = pack2(sA[ar][kb + 16 + 2*j], sA[ar][kb + 16 + 2*j + 1]);
    }
    #pragma unroll
    for (int t = 0; t < 4; ++t) {
      FragAB fb;
      const int bn = wc + t * 16 + l16;
      #pragma unroll
      for (int j = 0; j < 4; ++j) {
        fb.u[j]     = pack2(sB[bn][kb + 2*j],      sB[bn][kb + 2*j + 1]);
        fb.u[4 + j] = pack2(sB[bn][kb + 16 + 2*j], sB[bn][kb + 16 + 2*j + 1]);
      }
      acc[t] = __builtin_amdgcn_wmma_f32_16x16x32_bf16(false, fa.b, false, fb.b,
                                                       (short)0, acc[t], false, false);
    }
    __syncthreads();
  }

  // ---- epilogue: C layout VGPR i -> row (i | i+8), col = lane&15 ----
  const bool hb = (flags & GEMM_BIAS) != 0;
  const bool hr = (flags & GEMM_RELU) != 0;
  #pragma unroll
  for (int t = 0; t < 4; ++t) {
    int col = n0 + wc + t * 16 + l16;
    if (col >= N) continue;
    float bv = hb ? bias[col] : 0.0f;
    #pragma unroll
    for (int i = 0; i < 8; ++i) {
      int row = m0 + wr + i + (half ? 8 : 0);
      if (row >= M) continue;
      float v = (acc[t][i] + bv) * alpha;
      if (hr) v = fmaxf(v, 0.0f);
      C[(long)row * ldc + col] = v;
    }
  }
}

// ------------------------------ LayerNorm (D=256) ---------------------------
__global__ __launch_bounds__(256) void ln_kernel(
    const float* __restrict__ x, const float* __restrict__ res,
    const float* __restrict__ g, const float* __restrict__ b,
    float* __restrict__ y)
{
  long row = blockIdx.x;
  int  d   = threadIdx.x;
  float v = x[row * 256 + d];
  if (res) v += res[row * 256 + d];
  float mean = blk_sum(v) * (1.0f / 256.0f);
  float diff = v - mean;
  float var  = blk_sum(diff * diff) * (1.0f / 256.0f);
  y[row * 256 + d] = diff * rsqrtf(var + 1e-5f) * g[d] + b[d];
}

// ------------------------------ softmax over 256 cols (in place) ------------
// maskmode 0: none; 1: +mask[col]*FP16_MIN; 2: +(1-mask[(row%256)*256+col])*FP16_MIN
__global__ __launch_bounds__(256) void softmax256_kernel(
    float* __restrict__ x, const float* __restrict__ mask, int maskmode)
{
  long row = blockIdx.x;
  int  c   = threadIdx.x;
  float v = x[row * 256 + c];
  if (maskmode == 1)      v += mask[c] * FP16_MIN;
  else if (maskmode == 2) v += (1.0f - mask[(row & 255) * 256 + c]) * FP16_MIN;
  float m = blk_max(v);
  float e = __expf(v - m);
  float s = blk_sum(e);
  x[row * 256 + c] = e / s;
}

// ------------------------------ transposed softmax over SV ------------------
// taw[h][t][v] = softmax_v( aw[h][v][t] + kpm[v]*FP16_MIN )
__global__ __launch_bounds__(256) void softmax_taw_kernel(
    const float* __restrict__ aw, const float* __restrict__ kpm,
    float* __restrict__ taw, int sv)
{
  int t = blockIdx.x, h = blockIdx.y;
  const float* src = aw + (long)h * sv * 256 + t;
  float* dst = taw + ((long)h * 256 + t) * (long)sv;
  float m = -3.4e38f;
  for (int v = threadIdx.x; v < sv; v += 256)
    m = fmaxf(m, src[(long)v * 256] + kpm[v] * FP16_MIN);
  m = blk_max(m);
  float s = 0.0f;
  for (int v = threadIdx.x; v < sv; v += 256) {
    float e = __expf(src[(long)v * 256] + kpm[v] * FP16_MIN - m);
    dst[v] = e;
    s += e;
  }
  s = blk_sum(s);
  float inv = 1.0f / s;
  for (int v = threadIdx.x; v < sv; v += 256) dst[v] *= inv;
}

// ------------------------------ misc elementwise ----------------------------
__global__ void tpos_kernel(float* __restrict__ out) {
  int t = blockIdx.x, d = threadIdx.x;
  float dimt = __powf(10000.0f, 2.0f * (float)(d / 2) / 256.0f);
  float x = (float)t * 6.28318530717958647692f / dimt;
  out[t * 256 + d] = (d & 1) ? __cosf(x) : __sinf(x);
}

__global__ void add_kernel(const float* __restrict__ a, const float* __restrict__ b,
                           float* __restrict__ y, long n) {
  long i = (long)blockIdx.x * blockDim.x + threadIdx.x;
  if (i < n) y[i] = a[i] + b[i];
}

__global__ void res_gamma_kernel(const float* __restrict__ base,
                                 const float* __restrict__ gamma,
                                 const float* __restrict__ delta,
                                 float* __restrict__ y, long n) {
  long i = (long)blockIdx.x * blockDim.x + threadIdx.x;
  if (i < n) y[i] = base[i] + gamma[i & 255] * delta[i];
}

// ------------------------------ deformable attention gather -----------------
// Block = one query v; 8 waves = 8 heads; 32 lanes = 32 channels.
__global__ __launch_bounds__(256) void deform_kernel(
    const float* __restrict__ value, const float* __restrict__ offl,
    const float* __restrict__ awl, const float* __restrict__ ref,
    const float* __restrict__ kpm, float* __restrict__ out)
{
  int v = blockIdx.x;
  int h = threadIdx.x >> 5;
  int c = threadIdx.x & 31;

  const float* alog = awl + (long)v * 128 + h * 16;
  float mx = -3.4e38f;
  #pragma unroll
  for (int i = 0; i < 16; ++i) mx = fmaxf(mx, alog[i]);
  float a16[16]; float s = 0.0f;
  #pragma unroll
  for (int i = 0; i < 16; ++i) { a16[i] = __expf(alog[i] - mx); s += a16[i]; }
  float inv = 1.0f / s;

  const float* off = offl + (long)v * 256 + h * 32;
  const int LH[4] = {84, 42, 21, 11};
  const int LW[4] = {84, 42, 21, 11};
  const int LS[4] = {0, 7056, 8820, 9261};

  float acc = 0.0f;
  #pragma unroll
  for (int l = 0; l < 4; ++l) {
    float Wl = (float)LW[l], Hl = (float)LH[l];
    float rx = ref[(long)v * 8 + l * 2 + 0];
    float ry = ref[(long)v * 8 + l * 2 + 1];
    #pragma unroll
    for (int p = 0; p < 4; ++p) {
      float lx = rx + off[l * 8 + p * 2 + 0] / Wl;
      float ly = ry + off[l * 8 + p * 2 + 1] / Hl;
      float x = lx * Wl - 0.5f;
      float y = ly * Hl - 0.5f;
      float xf = floorf(x), yf = floorf(y);
      int x0 = (int)xf, y0 = (int)yf;
      float wx1 = x - xf, wx0 = 1.0f - wx1;
      float wy1 = y - yf, wy0 = 1.0f - wy1;
      float sval = 0.0f;
      #pragma unroll
      for (int tap = 0; tap < 4; ++tap) {
        int xi = x0 + (tap & 1), yi = y0 + (tap >> 1);
        float wt = ((tap & 1) ? wx1 : wx0) * ((tap >> 1) ? wy1 : wy0);
        if (xi >= 0 && xi < LW[l] && yi >= 0 && yi < LH[l]) {
          long rowid = LS[l] + (long)yi * LW[l] + xi;
          sval += wt * value[rowid * 256 + h * 32 + c] * (1.0f - kpm[rowid]);
        }
      }
      acc += a16[l * 4 + p] * inv * sval;
    }
  }
  out[(long)v * 256 + h * 32 + c] = acc;
}

// ---------------------------------------------------------------------------
static inline void gemm(hipStream_t s, const float* A, const float* B,
                        const float* bias, float* C,
                        int M, int N, int K, int lda, int ldb, int ldc,
                        long sA, long sB, long sC, int batch,
                        float alpha, int flags)
{
  dim3 g((N + 127) / 128, (M + 63) / 64, batch);
  gemm_bf16_wmma<<<g, 256, 0, s>>>(A, B, bias, C, M, N, K, lda, ldb, ldc,
                                   sA, sB, sC, alpha, flags);
}

extern "C" void kernel_launch(void* const* d_in, const int* in_sizes, int n_in,
                              void* d_out, int out_size, void* d_ws, size_t ws_size,
                              hipStream_t stream)
{
  (void)in_sizes; (void)n_in; (void)out_size; (void)ws_size;

  const int   SV  = SV_CONST;
  const long  SVD = (long)SV * 256;      // SV x D
  const long  SV4 = (long)SV * 1024;     // SV x EF
  const long  TD  = 256L * 256;          // ST x D
  const long  T4  = 256L * 1024;         // ST x EF
  const float SCALE_F = 0.0625f;         // (1024/4)^(-0.5)
  const float SCALE_T = 0.125f;          // 1/sqrt(64)

  const float* vf    = (const float*)d_in[0];
  const float* vpe   = (const float*)d_in[1];
  const float* kpm   = (const float*)d_in[4];
  const float* ref   = (const float*)d_in[5];
  const float* tf    = (const float*)d_in[6];
  const float* tmask = (const float*)d_in[7];
  const float* tsam  = (const float*)d_in[8];
  #define PAR(i) ((const float*)d_in[9 + (i)])
  // lins order: vproj tproj vval tval out_v out_t te_q te_k te_v te_o te_fc1
  //             te_fc2 val_proj samp_off attn_w out_proj d_fc1 d_fc2
  const float *vproj_w = PAR(0),  *vproj_b = PAR(1);
  const float *tproj_w = PAR(2),  *tproj_b = PAR(3);
  const float *vval_w  = PAR(4),  *vval_b  = PAR(5);
  const float *tval_w  = PAR(6),  *tval_b  = PAR(7);
  const float *out_v_w = PAR(8),  *out_v_b = PAR(9);
  const float *out_t_w = PAR(10), *out_t_b = PAR(11);
  const float *te_q_w  = PAR(12), *te_q_b  = PAR(13);
  const float *te_k_w  = PAR(14), *te_k_b  = PAR(15);
  const float *te_v_w  = PAR(16), *te_v_b  = PAR(17);
  const float *te_o_w  = PAR(18), *te_o_b  = PAR(19);
  const float *te_fc1_w= PAR(20), *te_fc1_b= PAR(21);
  const float *te_fc2_w= PAR(22), *te_fc2_b= PAR(23);
  const float *valp_w  = PAR(24), *valp_b  = PAR(25);
  const float *soff_w  = PAR(26), *soff_b  = PAR(27);
  const float *attw_w  = PAR(28), *attw_b  = PAR(29);
  const float *outp_w  = PAR(30), *outp_b  = PAR(31);
  const float *dfc1_w  = PAR(32), *dfc1_b  = PAR(33);
  const float *dfc2_w  = PAR(34), *dfc2_b  = PAR(35);
  const float *ln_v_g  = PAR(36), *ln_v_b  = PAR(37);
  const float *ln_t_g  = PAR(38), *ln_t_b  = PAR(39);
  const float *teln1_g = PAR(40), *teln1_b = PAR(41);
  const float *teln2_g = PAR(42), *teln2_b = PAR(43);
  const float *dln1_g  = PAR(44), *dln1_b  = PAR(45);
  const float *dln2_g  = PAR(46), *dln2_b  = PAR(47);
  const float *gamma_v = PAR(48), *gamma_t = PAR(49);

  float* W = (float*)d_ws;
  size_t off = 0;
  auto alloc = [&](long n) { float* p = W + off; off += (size_t)n; return p; };

  float* tpos   = alloc(TD);
  float* vn     = alloc(SVD);
  float* tn     = alloc(TD);
  float* q      = alloc(SV4);   // reused later as vmerge
  float* kbuf   = alloc(T4);
  float* vvbuf  = alloc(SV4);   // reused later as hs/value/offl/sampled
  float* tv     = alloc(T4);
  float* aw     = alloc(SV4);   // aw+taw contiguous -> reused as FFN hidden f1
  float* taw    = alloc(SV4);
  float* tmerge = alloc(T4);
  float* dv     = alloc(SVD);   // reused later as attn projection
  float* dt     = alloc(TD);
  float* vres   = alloc(SVD);
  float* tres   = alloc(TD);
  float* qk     = alloc(TD);
  float* qh     = alloc(TD);
  float* kh     = alloc(TD);
  float* vh     = alloc(TD);
  float* scores = alloc(T4);
  float* ctx    = alloc(TD);
  float* ctx2   = alloc(TD);
  float* tln    = alloc(TD);
  float* th1    = alloc(T4);
  float* th2    = alloc(TD);
  float* awlbuf = alloc((long)SV * 128);
  float* vln    = alloc(SVD);
  float* f2     = alloc(SVD);

  // aliases (lifetimes verified: previous contents dead before reuse)
  float* vmerge   = q;
  float* hs       = vvbuf;
  float* value    = vvbuf + SVD;
  float* offl     = vvbuf + 2 * SVD;
  float* sampled  = vvbuf + 3 * SVD;
  float* f1       = aw;            // SV x 2048 spans aw+taw
  float* attnproj = dv;

  float* v_out = (float*)d_out;           // SV x 256
  float* t_out = (float*)d_out + SVD;     // 256 x 256

  // ---- pos embed + layernorms ----
  tpos_kernel<<<256, 256, 0, stream>>>(tpos);
  ln_kernel<<<SV, 256, 0, stream>>>(vf, nullptr, ln_v_g, ln_v_b, vn);
  ln_kernel<<<256, 256, 0, stream>>>(tf, nullptr, ln_t_g, ln_t_b, tn);

  // ---- bi-directional MHA projections ----
  gemm(stream, vn, vproj_w, vproj_b, q,   SV, 1024, 256, 256, 1024, 1024, 0,0,0,1, SCALE_F, GEMM_BIAS);
  gemm(stream, tn, tproj_w, tproj_b, kbuf,256, 1024, 256, 256, 1024, 1024, 0,0,0,1, 1.0f, GEMM_BIAS);
  gemm(stream, vn, vval_w,  vval_b,  vvbuf,SV,1024, 256, 256, 1024, 1024, 0,0,0,1, 1.0f, GEMM_BIAS);
  gemm(stream, tn, tval_w,  tval_b,  tv,  256, 1024, 256, 256, 1024, 1024, 0,0,0,1, 1.0f, GEMM_BIAS);

  // aw[h] = q_h (SVx256) @ k_h^T (256x256), batched over 4 heads
  gemm(stream, q, kbuf, nullptr, aw, SV, 256, 256, 1024, 1024, 256,
       256, 256, (long)SV * 256, 4, 1.0f, GEMM_TRANSB);

  // softmaxes
  softmax_taw_kernel<<<dim3(256, 4), 256, 0, stream>>>(aw, kpm, taw, SV);
  softmax256_kernel<<<4 * SV, 256, 0, stream>>>(aw, tmask, 1);   // vaw in place

  // vmerge[h] = vaw_h @ tv_h ; tmerge[h] = taw_h @ vv_h
  gemm(stream, aw,  tv,    nullptr, vmerge, SV, 256, 256, 256, 1024, 1024,
       (long)SV * 256, 256, 256, 4, 1.0f, 0);
  gemm(stream, taw, vvbuf, nullptr, tmerge, 256, 256, SV, SV, 1024, 1024,
       256L * SV, 256, 256, 4, 1.0f, 0);

  gemm(stream, vmerge, out_v_w, out_v_b, dv, SV, 256, 1024, 1024, 256, 256, 0,0,0,1, 1.0f, GEMM_BIAS);
  gemm(stream, tmerge, out_t_w, out_t_b, dt, 256, 256, 1024, 1024, 256, 256, 0,0,0,1, 1.0f, GEMM_BIAS);

  res_gamma_kernel<<<(int)((SVD + 255) / 256), 256, 0, stream>>>(vn, gamma_v, dv, vres, SVD);
  res_gamma_kernel<<<(int)((TD + 255) / 256), 256, 0, stream>>>(tn, gamma_t, dt, tres, TD);

  // ---- text self-attention ----
  add_kernel<<<(int)((TD + 255) / 256), 256, 0, stream>>>(tres, tpos, qk, TD);
  gemm(stream, qk,   te_q_w, te_q_b, qh, 256, 256, 256, 256, 256, 256, 0,0,0,1, 1.0f, GEMM_BIAS);
  gemm(stream, qk,   te_k_w, te_k_b, kh, 256, 256, 256, 256, 256, 256, 0,0,0,1, 1.0f, GEMM_BIAS);
  gemm(stream, tres, te_v_w, te_v_b, vh, 256, 256, 256, 256, 256, 256, 0,0,0,1, 1.0f, GEMM_BIAS);

  gemm(stream, qh, kh, nullptr, scores, 256, 256, 64, 256, 256, 256,
       64, 64, 256L * 256, 4, SCALE_T, GEMM_TRANSB);
  softmax256_kernel<<<1024, 256, 0, stream>>>(scores, tsam, 2);
  gemm(stream, scores, vh, nullptr, ctx, 256, 64, 256, 256, 256, 256,
       256L * 256, 64, 64, 4, 1.0f, 0);
  gemm(stream, ctx, te_o_w, te_o_b, ctx2, 256, 256, 256, 256, 256, 256, 0,0,0,1, 1.0f, GEMM_BIAS);

  ln_kernel<<<256, 256, 0, stream>>>(tres, ctx2, teln1_g, teln1_b, tln);
  gemm(stream, tln, te_fc1_w, te_fc1_b, th1, 256, 1024, 256, 256, 1024, 1024, 0,0,0,1, 1.0f, GEMM_BIAS | GEMM_RELU);
  gemm(stream, th1, te_fc2_w, te_fc2_b, th2, 256, 256, 1024, 1024, 256, 256, 0,0,0,1, 1.0f, GEMM_BIAS);
  ln_kernel<<<256, 256, 0, stream>>>(tln, th2, teln2_g, teln2_b, t_out);

  // ---- deformable attention (vision) ----
  add_kernel<<<(int)((SVD + 255) / 256), 256, 0, stream>>>(vres, vpe, hs, SVD);
  gemm(stream, vres, valp_w, valp_b, value, SV, 256, 256, 256, 256, 256, 0,0,0,1, 1.0f, GEMM_BIAS);
  gemm(stream, hs,   soff_w, soff_b, offl,  SV, 256, 256, 256, 256, 256, 0,0,0,1, 1.0f, GEMM_BIAS);
  gemm(stream, hs,   attw_w, attw_b, awlbuf,SV, 128, 256, 256, 128, 128, 0,0,0,1, 1.0f, GEMM_BIAS);
  deform_kernel<<<SV, 256, 0, stream>>>(value, offl, awlbuf, ref, kpm, sampled);
  gemm(stream, sampled, outp_w, outp_b, attnproj, SV, 256, 256, 256, 256, 256, 0,0,0,1, 1.0f, GEMM_BIAS);

  ln_kernel<<<SV, 256, 0, stream>>>(vres, attnproj, dln1_g, dln1_b, vln);
  gemm(stream, vln, dfc1_w, dfc1_b, f1, SV, 2048, 256, 256, 2048, 2048, 0,0,0,1, 1.0f, GEMM_BIAS | GEMM_RELU);
  gemm(stream, f1,  dfc2_w, dfc2_b, f2, SV, 256, 2048, 2048, 256, 256, 0,0,0,1, 1.0f, GEMM_BIAS);
  ln_kernel<<<SV, 256, 0, stream>>>(vln, f2, dln2_g, dln2_b, v_out);
}
